// Decoder_58626303590489
// MI455X (gfx1250) — matrix operand order repaired and checked by reading, not compiled, not run on hardware
//
#include <hip/hip_runtime.h>

typedef unsigned int u32;
typedef unsigned short u16;
typedef unsigned long long u64;

typedef __attribute__((ext_vector_type(16))) __bf16 v16bf;
typedef __attribute__((ext_vector_type(8)))  float  v8f;
typedef __attribute__((ext_vector_type(4)))  u32    u32x4;

union FragBF {
  v16bf v;
  u32   u[8];
  u32x4 q4[2];
  u16   h[16];
};

__device__ __forceinline__ u16 f2bf(float f) {
  u32 x = __float_as_uint(f);
  u32 r = x + 0x7FFFu + ((x >> 16) & 1u);   // RNE
  return (u16)(r >> 16);
}

__device__ __forceinline__ v8f v8f_zero() {
  v8f z;
  #pragma unroll
  for (int i = 0; i < 8; ++i) z[i] = 0.0f;
  return z;
}

__device__ __forceinline__ u32 mix32(u32 a, u32 b) {
  u32 h = a * 0x9E3779B1u ^ (b + 0x85EBCA77u + (a << 6) + (a >> 2));
  h ^= h >> 16; h *= 0x7FEB352Du; h ^= h >> 15; h *= 0x846CA68Bu; h ^= h >> 16;
  return h;
}

// LDS byte offset of a __shared__ object (flat -> AS(3) addrspacecast strips base)
__device__ __forceinline__ u32 lds_off(const void* p) {
  return (u32)(size_t)(__attribute__((address_space(3))) const void*)p;
}

// CDNA5 async global->LDS copy, 16B per lane, GV mode (CDNA5 ISA).
// Tracked with ASYNCcnt.
__device__ __forceinline__ void async_g2l_b128(u32 lds_byte_off, const void* gptr) {
  asm volatile("global_load_async_to_lds_b128 %0, %1, off"
               :: "v"(lds_byte_off), "v"((u64)(size_t)gptr)
               : "memory");
}
__device__ __forceinline__ void wait_asynccnt0() {
  asm volatile("s_wait_asynccnt 0x0" ::: "memory");
}
__device__ __forceinline__ void wait_asynccnt2() {
  // steady-state pipeline: allow the 2 just-issued (next-tile) copies to remain
  asm volatile("s_wait_asynccnt 0x2" ::: "memory");
}

// ---------------------------------------------------------------------------
// Generic bf16 WMMA GEMM: C[M,N](f32 or bf16) = A[M,K] * B[K,N] (+bias)(+relu)
// Block tile 128x128, 8 waves (wave w -> rows 16w..16w+15, all 128 cols).
// Software-pipelined K loop: tile t+1 staged via GLOBAL_LOAD_ASYNC_TO_LDS_B128
// into the alternate LDS buffer while tile t feeds the WMMA burst.
// flags: bit0 = relu, bit1 = bf16 output
// ---------------------------------------------------------------------------
#define TILE_M 128
#define TILE_N 128
#define TILE_K 32
#define AST 40   // LDS row stride (halves) for A tile  (80B, 16B aligned)
#define BST 40   // LDS row stride (halves) for Bt tile

__global__ __launch_bounds__(256)
void gemm_bf16_wmma(const u16* __restrict__ A, int lda,
                    const u16* __restrict__ B, int ldb,
                    void* __restrict__ Cout, int ldc, int K,
                    const float* __restrict__ bias, int flags) {
  __shared__ u16 As[2][TILE_M * AST];
  __shared__ u16 Bt[2][TILE_N * BST];
  const int tid  = threadIdx.x;
  const int lane = tid & 31;
  const int wid  = tid >> 5;
  const int m0   = blockIdx.y * TILE_M;
  const int n0   = blockIdx.x * TILE_N;
  const int l16  = lane & 15;
  const int lh   = (lane < 16) ? 0 : 1;

  v8f acc[8];
  #pragma unroll
  for (int i = 0; i < 8; ++i) acc[i] = v8f_zero();

  const int arow = tid >> 1;
  const int acol = (tid & 1) << 4;
  const int brow = tid >> 3;
  const int bcol = (tid & 7) << 4;

  // Stage one K-tile: A via async global->LDS (2x16B/thread, no VGPR roundtrip),
  // B transposed into Bt[n][k] via regular stores (DScnt, drained at the barrier).
  auto stage_tile = [&](int k0, int buf) {
    const u16* ga = A + (size_t)(m0 + arow) * lda + k0 + acol;
    const u32  la = lds_off(&As[buf][arow * AST + acol]);
    async_g2l_b128(la,      ga);
    async_g2l_b128(la + 16, ga + 8);
    const u32x4* src = (const u32x4*)(B + (size_t)(k0 + brow) * ldb + n0 + bcol);
    u32x4 b0 = src[0];
    u32x4 b1 = src[1];
    u16 tmp[16];
    *(u32x4*)&tmp[0] = b0;
    *(u32x4*)&tmp[8] = b1;
    #pragma unroll
    for (int i = 0; i < 16; ++i) Bt[buf][(bcol + i) * BST + brow] = tmp[i];
  };

  const int ntile = K / TILE_K;
  stage_tile(0, 0);

  for (int t = 0; t < ntile; ++t) {
    const int cur = t & 1;
    if (t + 1 < ntile) {
      stage_tile((t + 1) * TILE_K, cur ^ 1);   // next tile streams behind compute
      wait_asynccnt2();                        // tile t complete; t+1 in flight
    } else {
      wait_asynccnt0();
    }
    if (t + 2 < ntile) {                       // pull t+2 toward the WGP (L2->WGP$)
      __builtin_prefetch(A + (size_t)(m0 + arow) * lda + (t + 2) * TILE_K + acol, 0, 3);
      __builtin_prefetch(B + (size_t)((t + 2) * TILE_K + brow) * ldb + n0 + bcol, 0, 3);
    }
    __syncthreads();

    // A fragment (16x32 bf16, ISA layout: lanes<16 K 0..7,16..23; lanes>=16 K 8..15,24..31)
    FragBF af;
    {
      const int r = (wid << 4) + l16;
      const int base = lh << 3;
      #pragma unroll
      for (int v = 0; v < 8; ++v) {
        const int kk = (v < 4) ? (base + 2 * v) : (16 + base + 2 * (v - 4));
        af.u[v] = *(const u32*)&As[cur][r * AST + kk];
      }
    }
    #pragma unroll
    for (int nt = 0; nt < 8; ++nt) {
      FragBF bf;
      const int n  = (nt << 4) + l16;
      const int kb = lh << 3;   // lanes<16: K 0..7,16..23, lanes>=16: K 8..15,24..31
      bf.q4[0] = *(const u32x4*)&Bt[cur][n * BST + kb];
      bf.q4[1] = *(const u32x4*)&Bt[cur][n * BST + kb + 16];
      acc[nt] = __builtin_amdgcn_wmma_f32_16x16x32_bf16(false, af.v, false, bf.v,
                                                        (short)0, acc[nt], false, false);
    }
    __syncthreads();   // all waves done reading buffer `cur` before it is restaged
  }

  const int rbase = (wid << 4) + (lh << 3);
  #pragma unroll
  for (int nt = 0; nt < 8; ++nt) {
    const int n = n0 + (nt << 4) + l16;
    const float bb = bias ? bias[n] : 0.0f;
    #pragma unroll
    for (int v = 0; v < 8; ++v) {
      float c = acc[nt][v] + bb;
      if (flags & 1) c = fmaxf(c, 0.0f);
      const size_t off = (size_t)(m0 + rbase + v) * ldc + n;
      if (flags & 2) ((u16*)Cout)[off] = f2bf(c);
      else           ((float*)Cout)[off] = c;
    }
  }
}

// ---------------------------------------------------------------------------
// Flash-attention: one wave per (16-query tile, head). qkv bf16 [2048,1536]
// layout [n][ q(0..511) | k(512..1023) | v(1024..1535) ], head dim 64.
// ---------------------------------------------------------------------------
__global__ __launch_bounds__(32)
void attn_fa(const u16* __restrict__ qkv, u16* __restrict__ outbf) {
  __shared__ u16 Pt[16 * 34];
  __shared__ u16 Vt[64 * BST];
  const int lane = threadIdx.x;
  const int head = blockIdx.y;
  const int q0   = blockIdx.x << 4;
  const int hoff = head << 6;
  const int l16  = lane & 15;
  const int lh   = (lane < 16) ? 0 : 1;

  FragBF qf[2];
  {
    const u16* Qrow = qkv + (size_t)(q0 + l16) * 1536 + hoff;
    const int base = lh << 3;
    #pragma unroll
    for (int dc = 0; dc < 2; ++dc) {
      #pragma unroll
      for (int v = 0; v < 8; ++v) {
        const int kk = dc * 32 + ((v < 4) ? (base + 2 * v) : (16 + base + 2 * (v - 4)));
        qf[dc].u[v] = *(const u32*)(Qrow + kk);
      }
    }
  }

  v8f o[4];
  #pragma unroll
  for (int i = 0; i < 4; ++i) o[i] = v8f_zero();
  float m[8], l[8];
  #pragma unroll
  for (int v = 0; v < 8; ++v) { m[v] = -1e30f; l[v] = 0.0f; }

  const u16* Kp = qkv + 512 + hoff;
  const u16* Vp = qkv + 1024 + hoff;

  for (int c = 0; c < 2048; c += 32) {
    // S = Q * K^T for 32 keys (two 16-key groups), contraction d=64 split in two
    v8f s[2];
    #pragma unroll
    for (int kg = 0; kg < 2; ++kg) {
      const u16* Krow = Kp + (size_t)(c + (kg << 4) + l16) * 1536;
      const int kb = lh << 3;
      FragBF b0, b1;
      b0.q4[0] = *(const u32x4*)(Krow + kb);
      b0.q4[1] = *(const u32x4*)(Krow + kb + 16);
      b1.q4[0] = *(const u32x4*)(Krow + 32 + kb);
      b1.q4[1] = *(const u32x4*)(Krow + 32 + kb + 16);
      v8f z = v8f_zero();
      z = __builtin_amdgcn_wmma_f32_16x16x32_bf16(false, qf[0].v, false, b0.v, (short)0, z, false, false);
      z = __builtin_amdgcn_wmma_f32_16x16x32_bf16(false, qf[1].v, false, b1.v, (short)0, z, false, false);
      s[kg] = z;
    }
    // online softmax; rows live in VGPR index v (+8 for upper lane half)
    float al[8];
    #pragma unroll
    for (int v = 0; v < 8; ++v) {
      float s0 = s[0][v] * 0.125f;   // 1/sqrt(64)
      float s1 = s[1][v] * 0.125f;
      float t = fmaxf(s0, s1);
      t = fmaxf(t, __shfl_xor(t, 1, 32));
      t = fmaxf(t, __shfl_xor(t, 2, 32));
      t = fmaxf(t, __shfl_xor(t, 4, 32));
      t = fmaxf(t, __shfl_xor(t, 8, 32));
      const float mn = fmaxf(m[v], t);
      al[v] = __expf(m[v] - mn);
      const float p0 = __expf(s0 - mn);
      const float p1 = __expf(s1 - mn);
      float rs = p0 + p1;
      rs += __shfl_xor(rs, 1, 32);
      rs += __shfl_xor(rs, 2, 32);
      rs += __shfl_xor(rs, 4, 32);
      rs += __shfl_xor(rs, 8, 32);
      l[v] = l[v] * al[v] + rs;
      m[v] = mn;
      const int prow = (v + (lh << 3)) * 34;
      Pt[prow + l16]      = f2bf(p0);   // C-layout -> LDS round trip to A-layout
      Pt[prow + 16 + l16] = f2bf(p1);
    }
    #pragma unroll
    for (int dt = 0; dt < 4; ++dt) {
      #pragma unroll
      for (int v = 0; v < 8; ++v) o[dt][v] *= al[v];
    }
    // V chunk transposed into LDS: lane <-> key c+lane
    {
      const u16* Vrow = Vp + (size_t)(c + lane) * 1536;
      #pragma unroll
      for (int d = 0; d < 64; ++d) Vt[d * BST + lane] = Vrow[d];
    }
    __syncthreads();

    FragBF pf;
    {
      const int base = lh << 3;
      #pragma unroll
      for (int v = 0; v < 8; ++v) {
        const int kk = (v < 4) ? (base + 2 * v) : (16 + base + 2 * (v - 4));
        pf.u[v] = *(const u32*)&Pt[l16 * 34 + kk];
      }
    }
    #pragma unroll
    for (int dt = 0; dt < 4; ++dt) {
      FragBF vf;
      const int d  = (dt << 4) + l16;
      const int kb = lh << 3;
      vf.q4[0] = *(const u32x4*)&Vt[d * BST + kb];
      vf.q4[1] = *(const u32x4*)&Vt[d * BST + kb + 16];
      o[dt] = __builtin_amdgcn_wmma_f32_16x16x32_bf16(false, pf.v, false, vf.v,
                                                      (short)0, o[dt], false, false);
    }
    __syncthreads();
  }

  #pragma unroll
  for (int dt = 0; dt < 4; ++dt) {
    #pragma unroll
    for (int v = 0; v < 8; ++v) {
      const float val = o[dt][v] / l[v];
      outbf[(size_t)(q0 + v + (lh << 3)) * 512 + hoff + (dt << 4) + l16] = f2bf(val);
    }
  }
}

// ---------------------------------------------------------------------------
// LayerNorm(h + t) with residual; writes f32 + bf16 copies. One row per block.
// ---------------------------------------------------------------------------
__global__ __launch_bounds__(256)
void ln_residual(const float* __restrict__ hin, const float* __restrict__ t,
                 const float* __restrict__ g, const float* __restrict__ b,
                 float* __restrict__ hout, u16* __restrict__ hbf) {
  __shared__ float rs[256], rq[256];
  const int r = blockIdx.x;
  const int tid = threadIdx.x;
  const size_t base = (size_t)r * 512;
  const float x0 = hin[base + tid] + t[base + tid];
  const float x1 = hin[base + 256 + tid] + t[base + 256 + tid];
  rs[tid] = x0 + x1;
  rq[tid] = x0 * x0 + x1 * x1;
  __syncthreads();
  for (int s = 128; s > 0; s >>= 1) {
    if (tid < s) { rs[tid] += rs[tid + s]; rq[tid] += rq[tid + s]; }
    __syncthreads();
  }
  const float mean = rs[0] * (1.0f / 512.0f);
  const float var  = rq[0] * (1.0f / 512.0f) - mean * mean;
  const float rstd = rsqrtf(var + 1e-5f);
  const float y0 = (x0 - mean) * rstd * g[tid] + b[tid];
  const float y1 = (x1 - mean) * rstd * g[256 + tid] + b[256 + tid];
  hout[base + tid] = y0;        hbf[base + tid] = f2bf(y0);
  hout[base + 256 + tid] = y1;  hbf[base + 256 + tid] = f2bf(y1);
}

__global__ __launch_bounds__(256)
void embed_kernel(const int* __restrict__ pts, const float* __restrict__ Wx,
                  const float* __restrict__ bx, float* __restrict__ h,
                  u16* __restrict__ hbf) {
  const int r = blockIdx.x;
  const float p0 = (float)pts[r * 4 + 0] * (1.0f / 70.0f);
  const float p1 = (float)pts[r * 4 + 1] * (1.0f / 70.0f);
  const float p2 = (float)pts[r * 4 + 2] * (1.0f / 70.0f);
  const float p3 = (float)pts[r * 4 + 3] * (1.0f / 70.0f);
  for (int j = threadIdx.x; j < 512; j += 256) {
    const float v = bx[j] + p0 * Wx[j] + p1 * Wx[512 + j] + p2 * Wx[1024 + j] + p3 * Wx[1536 + j];
    h[(size_t)r * 512 + j] = v;
    hbf[(size_t)r * 512 + j] = f2bf(v);
  }
}

__global__ __launch_bounds__(256)
void cvt_bf16(const float* __restrict__ in, u16* __restrict__ out, int n) {
  const int i = blockIdx.x * 256 + threadIdx.x;
  if (i < n) out[i] = f2bf(in[i]);
}

__global__ __launch_bounds__(256)
void col_mean(const float* __restrict__ h, float* __restrict__ mean) {
  const int c = blockIdx.x * 256 + threadIdx.x;
  float s = 0.0f;
  for (int r = 0; r < 2048; ++r) s += h[(size_t)r * 512 + c];
  mean[c] = s * (1.0f / 2048.0f);
}

__global__ __launch_bounds__(256)
void matvec(const float* __restrict__ x, const float* __restrict__ W,
            const float* __restrict__ bias, float* __restrict__ y, int K, int N) {
  const int j = blockIdx.x * 256 + threadIdx.x;
  if (j >= N) return;
  float s = bias ? bias[j] : 0.0f;
  for (int k = 0; k < K; ++k) s += x[k] * W[(size_t)k * N + j];
  y[j] = s;
}

// ---------------------------------------------------------------------------
// Sequential pointer decoder (512 dependent steps, 3 small kernels per step)
// ---------------------------------------------------------------------------
struct DecState { float lp, rw, sumexp; u64 pack; int forced, st; };

__global__ __launch_bounds__(256)
void dec_init(const float* __restrict__ costs, const float* __restrict__ hbar,
              const float* __restrict__ vq, int* __restrict__ mask,
              float* __restrict__ inith, float* __restrict__ query,
              DecState* __restrict__ sc) {
  const int tid = threadIdx.x;
  if (tid == 0) {
    int f = 0; float best = costs[0];
    for (int i = 1; i < 4; ++i) if (costs[i] < best) { best = costs[i]; f = i; }
    sc->forced = f; sc->st = f;
    sc->lp = 0.0f; sc->rw = 0.0f; sc->sumexp = 0.0f; sc->pack = 0ull;
  }
  for (int n = tid; n < 2048; n += 256) mask[n] = (n < 4) ? 1 : 0;
  for (int j = tid; j < 512; j += 256) { inith[j] = 0.0f; query[j] = hbar[j] + vq[j]; }
}

__global__ __launch_bounds__(256)
void dec_stepA(const float* __restrict__ query, const float* __restrict__ Wq,
               float* __restrict__ qW, DecState* __restrict__ sc) {
  const int tid = threadIdx.x;
  if (tid == 0) { sc->sumexp = 0.0f; sc->pack = 0ull; }
  for (int j = tid; j < 512; j += 256) {
    float s = 0.0f;
    for (int k = 0; k < 512; ++k) s += query[k] * Wq[(size_t)k * 512 + j];
    qW[j] = s;
  }
}

__global__ __launch_bounds__(256)
void dec_stepB(const float* __restrict__ kvec, const float* __restrict__ qW,
               const int* __restrict__ mask, float* __restrict__ logits,
               DecState* __restrict__ sc, int step) {
  const int n = blockIdx.x * 256 + threadIdx.x;
  const float* kr = kvec + (size_t)n * 512;
  float s = 0.0f;
  for (int k = 0; k < 512; ++k) s += kr[k] * qW[k];
  const float u = 10.0f * tanhf(s * 0.044194173824159216f);  // CLIP*tanh(./sqrt(512))
  const float lg = (mask[n] == 1) ? -1e9f : u;
  logits[n] = lg;
  atomicAdd(&sc->sumexp, __expf(lg));
  // gumbel-argmax categorical sample (hash RNG in lieu of threefry)
  const u32 hsh = mix32((u32)step * 2654435761u + 0x9E3779B9u, (u32)n);
  const float uu = ((float)hsh + 0.5f) * 2.3283064365386963e-10f;
  const float gl = lg + (-__logf(-__logf(uu)));
  u32 fb = __float_as_uint(gl);
  fb = (fb & 0x80000000u) ? ~fb : (fb | 0x80000000u);   // order-preserving float->uint
  const u64 pk = ((u64)fb << 32) | (u32)n;
  atomicMax(&sc->pack, pk);
}

__global__ __launch_bounds__(256)
void dec_stepC(const float* __restrict__ h, const float* __restrict__ hbar,
               const float* __restrict__ bv, const float* __restrict__ Wv,
               int* __restrict__ mask, const float* __restrict__ logits,
               float* __restrict__ inith, float* __restrict__ query,
               const float* __restrict__ costs, const int* __restrict__ pts,
               DecState* __restrict__ sc, float* __restrict__ out, int step) {
  __shared__ int sidx;
  const int tid = threadIdx.x;
  if (tid == 0) {
    const int idx = (step == 0) ? sc->forced : (int)(sc->pack & 0xFFFFFFFFull);
    const int st = sc->st;
    const float lp = sc->lp + (logits[idx] - __logf(sc->sumexp));
    const float dx = (float)pts[idx * 4 + 0] - (float)pts[st * 4 + 2];
    const float dy = (float)pts[idx * 4 + 1] - (float)pts[st * 4 + 3];
    const float ext = sqrtf(dx * dx + dy * dy);
    const float r = (ext + costs[st] + costs[idx]) * (1.0f / 70.0f);
    const float rw = sc->rw + ((step > 0) ? r : 0.0f);
    sc->lp = lp; sc->rw = rw; sc->st = idx;
    out[0] = lp; out[1] = rw; out[2 + step] = (float)idx;
    sidx = idx;
  }
  __syncthreads();
  const int idx = sidx;
  if (tid < 4) mask[((idx >> 2) << 2) + tid] = 1;
  if (tid == 0) mask[idx] = 1;
  if (step == 0) {
    for (int j = tid; j < 512; j += 256) inith[j] = h[(size_t)idx * 512 + j];
  }
  __syncthreads();
  const float* hr = h + (size_t)idx * 512;
  for (int j = tid; j < 512; j += 256) {
    float s = hbar[j] + bv[j];
    for (int k = 0; k < 512; ++k) s += inith[k] * Wv[(size_t)k * 512 + j];
    for (int k = 0; k < 512; ++k) s += hr[k] * Wv[(size_t)(512 + k) * 512 + j];
    query[j] = s;
  }
}

// ---------------------------------------------------------------------------
extern "C" void kernel_launch(void* const* d_in, const int* in_sizes, int n_in,
                              void* d_out, int out_size, void* d_ws, size_t ws_size,
                              hipStream_t stream) {
  (void)in_sizes; (void)n_in; (void)out_size; (void)ws_size;
  const int N = 2048, E = 512, E3 = 1536, L = 3;

  const int*   points = (const int*)d_in[1];
  const float* costs  = (const float*)d_in[2];
  const float* init_w = (const float*)d_in[4];
  const float* Wc     = (const float*)d_in[5];
  const float* bc     = (const float*)d_in[6];
  const float* Wv     = (const float*)d_in[7];
  const float* bvp    = (const float*)d_in[8];
  const float* Wx     = (const float*)d_in[9];
  const float* bx     = (const float*)d_in[10];
  const float* Wq     = (const float*)d_in[11];
  const float* Wk     = (const float*)d_in[12];
  const float* Wqkv   = (const float*)d_in[13];
  const float* Wo     = (const float*)d_in[14];
  const float* ln1g   = (const float*)d_in[15];
  const float* ln1b   = (const float*)d_in[16];
  const float* W1     = (const float*)d_in[17];
  const float* b1     = (const float*)d_in[18];
  const float* W2     = (const float*)d_in[19];
  const float* b2     = (const float*)d_in[20];
  const float* ln2g   = (const float*)d_in[21];
  const float* ln2b   = (const float*)d_in[22];

  char* p = (char*)d_ws;
  auto carve = [&](size_t bytes) -> void* {
    void* r = (void*)p; p += (bytes + 255) & ~(size_t)255; return r;
  };
  u16*   Wqkv_bf = (u16*)carve((size_t)L * E * E3 * 2);
  u16*   Wo_bf   = (u16*)carve((size_t)L * E * E * 2);
  u16*   W1_bf   = (u16*)carve((size_t)L * E * E * 2);
  u16*   W2_bf   = (u16*)carve((size_t)L * E * E * 2);
  u16*   Wk_bf   = (u16*)carve((size_t)E * E * 2);
  float* h       = (float*)carve((size_t)N * E * 4);
  u16*   h_bf    = (u16*)carve((size_t)N * E * 2);
  u16*   qkv_bf  = (u16*)carve((size_t)N * E3 * 2);
  u16*   ao_bf   = (u16*)carve((size_t)N * E * 2);
  float* tmpf    = (float*)carve((size_t)N * E * 4);
  u16*   ff_bf   = (u16*)carve((size_t)N * E * 2);
  float* kvecf   = (float*)carve((size_t)N * E * 4);
  float* meanv   = (float*)carve(E * 4);
  float* hbar    = (float*)carve(E * 4);
  float* vq      = (float*)carve(E * 4);
  float* query   = (float*)carve(E * 4);
  float* qWbuf   = (float*)carve(E * 4);
  float* inith   = (float*)carve(E * 4);
  float* logits  = (float*)carve(N * 4);
  int*   mask    = (int*)carve(N * 4);
  DecState* sc   = (DecState*)carve(sizeof(DecState));

  auto cvt = [&](const float* src, u16* dst, size_t n) {
    cvt_bf16<<<dim3((unsigned)((n + 255) / 256)), dim3(256), 0, stream>>>(src, dst, (int)n);
  };
  cvt(Wqkv, Wqkv_bf, (size_t)L * E * E3);
  cvt(Wo,   Wo_bf,   (size_t)L * E * E);
  cvt(W1,   W1_bf,   (size_t)L * E * E);
  cvt(W2,   W2_bf,   (size_t)L * E * E);
  cvt(Wk,   Wk_bf,   (size_t)E * E);

  embed_kernel<<<dim3(N), dim3(256), 0, stream>>>(points, Wx, bx, h, h_bf);

  for (int l = 0; l < L; ++l) {
    gemm_bf16_wmma<<<dim3(E3 / 128, N / 128), dim3(256), 0, stream>>>(
        h_bf, E, Wqkv_bf + (size_t)l * E * E3, E3, qkv_bf, E3, E, nullptr, 2);
    attn_fa<<<dim3(N / 16, 8), dim3(32), 0, stream>>>(qkv_bf, ao_bf);
    gemm_bf16_wmma<<<dim3(E / 128, N / 128), dim3(256), 0, stream>>>(
        ao_bf, E, Wo_bf + (size_t)l * E * E, E, tmpf, E, E, nullptr, 0);
    ln_residual<<<dim3(N), dim3(256), 0, stream>>>(h, tmpf, ln1g + l * E, ln1b + l * E, h, h_bf);
    gemm_bf16_wmma<<<dim3(E / 128, N / 128), dim3(256), 0, stream>>>(
        h_bf, E, W1_bf + (size_t)l * E * E, E, ff_bf, E, E, b1 + l * E, 3);
    gemm_bf16_wmma<<<dim3(E / 128, N / 128), dim3(256), 0, stream>>>(
        ff_bf, E, W2_bf + (size_t)l * E * E, E, tmpf, E, E, b2 + l * E, 0);
    ln_residual<<<dim3(N), dim3(256), 0, stream>>>(h, tmpf, ln2g + l * E, ln2b + l * E, h, h_bf);
  }

  gemm_bf16_wmma<<<dim3(E / 128, N / 128), dim3(256), 0, stream>>>(
      h_bf, E, Wk_bf, E, kvecf, E, E, nullptr, 0);
  col_mean<<<dim3(2), dim3(256), 0, stream>>>(h, meanv);
  matvec<<<dim3(2), dim3(256), 0, stream>>>(meanv, Wc, bc, hbar, 512, 512);
  matvec<<<dim3(2), dim3(256), 0, stream>>>(init_w, Wv, bvp, vq, 1024, 512);
  dec_init<<<dim3(1), dim3(256), 0, stream>>>(costs, hbar, vq, mask, inith, query, sc);

  for (int i = 0; i < 512; ++i) {
    dec_stepA<<<dim3(1), dim3(256), 0, stream>>>(query, Wq, qWbuf, sc);
    dec_stepB<<<dim3(8), dim3(256), 0, stream>>>(kvecf, qWbuf, mask, logits, sc, i);
    dec_stepC<<<dim3(1), dim3(256), 0, stream>>>(h, hbar, bvp, Wv, mask, logits, inith,
                                                 query, costs, points, sc, (float*)d_out, i);
  }
}
